// DeepseekV3MoE_42485816492039
// MI455X (gfx1250) — matrix operand order repaired and checked by reading, not compiled, not run on hardware
//
#include <hip/hip_runtime.h>

// ---------------- problem constants ----------------
#define HD    1024   // hidden
#define ID    704    // routed expert intermediate
#define NE    32     // experts
#define NG    8      // groups
#define TOPKG 3
#define TOPK  6
#define NT    2048   // tokens
#define CAPE  768    // per-expert capacity
#define ISH   1408   // shared expert intermediate (2*704)

typedef __bf16 v16bf __attribute__((ext_vector_type(16)));
typedef float  v8f   __attribute__((ext_vector_type(8)));
typedef unsigned int u32x4 __attribute__((ext_vector_type(4)));
typedef int          i32x8 __attribute__((ext_vector_type(8)));
typedef int          i32x4 __attribute__((ext_vector_type(4)));

#define HAVE_TDM __has_builtin(__builtin_amdgcn_tensor_load_to_lds)

union FragB { v16bf v; unsigned int u[8]; };

__device__ __forceinline__ int imin(int a, int b) { return a < b ? a : b; }

// pack two fp32 into one dword of bf16 {lo -> [15:0], hi -> [31:16]}
__device__ __forceinline__ unsigned int pack2bf(float lo, float hi) {
  unsigned int a = __builtin_bit_cast(unsigned int, lo) + 0x8000u;
  unsigned int b = __builtin_bit_cast(unsigned int, hi) + 0x8000u;
#if __has_builtin(__builtin_amdgcn_perm)
  return __builtin_amdgcn_perm(b, a, 0x07060302u);  // v_perm_b32: {b.hi16, a.hi16}
#else
  return (a >> 16) | (b & 0xFFFF0000u);
#endif
}
__device__ __forceinline__ unsigned short f2bf(float f) {
  unsigned int u = __builtin_bit_cast(unsigned int, f) + 0x8000u;
  return (unsigned short)(u >> 16);
}

__device__ __forceinline__ void wait_tensorcnt0() {
#if __has_builtin(__builtin_amdgcn_s_wait_tensorcnt)
  __builtin_amdgcn_s_wait_tensorcnt(0);
#else
  asm volatile("s_wait_tensorcnt 0x0" ::: "memory");
#endif
}

// =====================================================================
// Kernel 1: gating. One wave32 per token; lane e owns expert e.
// =====================================================================
__global__ __launch_bounds__(128)
void gate_kernel(const float* __restrict__ x, const float* __restrict__ gw,
                 int* __restrict__ tk_idx, float* __restrict__ tk_wt) {
  __shared__ float xs[4][HD];
  const int lane = threadIdx.x & 31;
  const int w    = threadIdx.x >> 5;
  const int t    = blockIdx.x * 4 + w;

  {
    const float4* xr  = (const float4*)(x + (long)t * HD);
    float4*       xsv = (float4*)xs[w];
    for (int i = lane; i < HD / 4; i += 32) xsv[i] = xr[i];
  }
  __syncthreads();

  const float4* g4 = (const float4*)(gw + (long)lane * HD);
  const float4* a4 = (const float4*)xs[w];
  float acc = 0.f;
  for (int i = 0; i < HD / 4; ++i) {
    float4 a = a4[i], b = g4[i];
    acc += a.x * b.x + a.y * b.y + a.z * b.z + a.w * b.w;
  }

  // softmax across 32 lanes (wave32)
  float mx = acc;
  for (int o = 16; o; o >>= 1) mx = fmaxf(mx, __shfl_xor(mx, o, 32));
  float p = __expf(acc - mx);
  float s = p;
  for (int o = 16; o; o >>= 1) s += __shfl_xor(s, o, 32);
  float prob = p / s;

  // max within each group of 4 consecutive lanes (E/G = 4)
  float gm = prob;
  gm = fmaxf(gm, __shfl_xor(gm, 1, 32));
  gm = fmaxf(gm, __shfl_xor(gm, 2, 32));

  float gs[NG];
#pragma unroll
  for (int j = 0; j < NG; ++j) gs[j] = __shfl(gm, j * 4, 32);
  unsigned gsel = 0;
#pragma unroll
  for (int it = 0; it < TOPKG; ++it) {
    float best = -1.f; int bi = 0;
#pragma unroll
    for (int j = 0; j < NG; ++j)
      if (!((gsel >> j) & 1u) && gs[j] > best) { best = gs[j]; bi = j; }
    gsel |= 1u << bi;
  }
  float msc = ((gsel >> (lane >> 2)) & 1u) ? prob : 0.f;

  float v[32];
#pragma unroll
  for (int j = 0; j < 32; ++j) v[j] = __shfl(msc, j, 32);
  if (lane == 0) {
    unsigned sel = 0;
    for (int kk = 0; kk < TOPK; ++kk) {
      float best = -1.f; int bi = 0;
#pragma unroll
      for (int j = 0; j < 32; ++j)
        if (!((sel >> j) & 1u) && v[j] > best) { best = v[j]; bi = j; }
      sel |= 1u << bi;
      tk_idx[t * TOPK + kk] = bi;
      tk_wt [t * TOPK + kk] = best;   // SCALE == 1.0
    }
  }
}

// =====================================================================
// Kernel 2/3: routing table build
// =====================================================================
__global__ void zero_counts_kernel(int* counts) {
  if (threadIdx.x < NE) counts[threadIdx.x] = 0;
}

__global__ __launch_bounds__(256)
void route_build_kernel(const int* __restrict__ tk_idx, const float* __restrict__ tk_wt,
                        int* __restrict__ counts, int* __restrict__ rows,
                        float* __restrict__ rwt) {
  int i = blockIdx.x * 256 + threadIdx.x;           // i < NT*TOPK (exact)
  int e = tk_idx[i];
  int pos = atomicAdd(&counts[e], 1);
  if (pos < CAPE) {
    rows[e * CAPE + pos] = i / TOPK;
    rwt [e * CAPE + pos] = tk_wt[i];
  }
}

// =====================================================================
// Kernel 4: fused gate+up GEMM + SiLU, bf16 WMMA, double-buffered LDS.
// Block = 256 thr (8 waves as 2M x 4N). Block tile 64(M) x 64(N).
// Wave tile 32(M) x 16(N): 2 A-frags vs {Bg,Bu} -> 4 WMMA per K-step.
// LDS pair layout: dword [row][K/2] holds bf16 {K, K+1}.
// =====================================================================
__global__ __launch_bounds__(256)
void gateup_kernel(const float* __restrict__ X,
                   const int* __restrict__ rowsAll,     // null => identity rows
                   const int* __restrict__ counts,      // null => nrowsFixed
                   int nrowsFixed,
                   const float* __restrict__ BgBase,    // (KDIM x N) per expert
                   const float* __restrict__ BuBase,
                   int N, long bStride,
                   unsigned short* __restrict__ actBase, // bf16 out (nrows x N)
                   long actStride, int rowsStride, int KDIM) {
  const int e = blockIdx.z;
  const float*    Bg    = BgBase  + (long)e * bStride;
  const float*    Bu    = BuBase  + (long)e * bStride;
  unsigned short* act   = actBase + (long)e * actStride;
  const int*      erows = rowsAll ? rowsAll + (long)e * rowsStride : nullptr;
  const int       nrows = counts ? imin(counts[e], rowsStride) : nrowsFixed;
  const int m0 = blockIdx.y * 64;
  if (m0 >= nrows) return;
  const int n0 = blockIdx.x * 64;

  __shared__ unsigned int sA [2][64][18];  // rows 72B -> 8B aligned
  __shared__ unsigned int sBg[2][64][18];
  __shared__ unsigned int sBu[2][64][18];

  const int tid  = threadIdx.x;
  const int lane = tid & 31, wave = tid >> 5;
  const int mi = wave >> 2, ni = wave & 3;
  const int l  = lane & 15, g = lane >> 4;

  // ---- per-thread staging assignment (constant over K loop) ----
  const int rA = tid >> 3, jA = tid & 7;
  const float* aSrc0 = nullptr;
  const float* aSrc1 = nullptr;
  {
    int m = m0 + rA;
    if (m < nrows) aSrc0 = X + (long)(erows ? erows[m] : m) * KDIM + jA * 4;
    m = m0 + rA + 32;
    if (m < nrows) aSrc1 = X + (long)(erows ? erows[m] : m) * KDIM + jA * 4;
  }
  const int nB = (tid & 15) * 4;
  const int kB = (tid >> 4) * 2;
  const int kp = kB >> 1;
  const float* bgSrc = Bg + (long)kB * N + n0 + nB;
  const float* buSrc = Bu + (long)kB * N + n0 + nB;

  auto stage = [&](int p, int kt) {
    float4 f0 = make_float4(0.f, 0.f, 0.f, 0.f);
    float4 f1 = make_float4(0.f, 0.f, 0.f, 0.f);
    if (aSrc0) f0 = *(const float4*)(aSrc0 + kt);
    if (aSrc1) f1 = *(const float4*)(aSrc1 + kt);
    uint2 d0; d0.x = pack2bf(f0.x, f0.y); d0.y = pack2bf(f0.z, f0.w);
    uint2 d1; d1.x = pack2bf(f1.x, f1.y); d1.y = pack2bf(f1.z, f1.w);
    *(uint2*)&sA[p][rA     ][jA * 2] = d0;
    *(uint2*)&sA[p][rA + 32][jA * 2] = d1;

    const float* r0 = bgSrc + (long)kt * N;
    float4 g0 = *(const float4*)r0;
    float4 g1 = *(const float4*)(r0 + N);
    const float* u0 = buSrc + (long)kt * N;
    float4 h0 = *(const float4*)u0;
    float4 h1 = *(const float4*)(u0 + N);
    sBg[p][nB + 0][kp] = pack2bf(g0.x, g1.x);
    sBg[p][nB + 1][kp] = pack2bf(g0.y, g1.y);
    sBg[p][nB + 2][kp] = pack2bf(g0.z, g1.z);
    sBg[p][nB + 3][kp] = pack2bf(g0.w, g1.w);
    sBu[p][nB + 0][kp] = pack2bf(h0.x, h1.x);
    sBu[p][nB + 1][kp] = pack2bf(h0.y, h1.y);
    sBu[p][nB + 2][kp] = pack2bf(h0.z, h1.z);
    sBu[p][nB + 3][kp] = pack2bf(h0.w, h1.w);
    if (kt + 64 < KDIM) {  // CDNA5 prefetch of the panel after next
      __builtin_prefetch(bgSrc + (long)(kt + 64) * N, 0, 1);
      __builtin_prefetch(buSrc + (long)(kt + 64) * N, 0, 1);
    }
  };

  v8f accg0 = {}, accg1 = {}, accu0 = {}, accu1 = {};
  stage(0, 0);
  int p = 0;
  for (int kt = 0; kt < KDIM; kt += 32, p ^= 1) {
    __syncthreads();
    if (kt + 32 < KDIM) stage(p ^ 1, kt + 32);  // overlap with WMMA on buf p

    FragB a0, a1, bg, bu;
    const int ar0 = mi * 32 + l;
#pragma unroll
    for (int v = 0; v < 4; ++v) {
      a0.u[v]     = sA[p][ar0     ][g * 4 + v];        // K = g*8 + 2v
      a0.u[v + 4] = sA[p][ar0     ][8 + g * 4 + v];    // K = 16 + g*8 + 2v
      a1.u[v]     = sA[p][ar0 + 16][g * 4 + v];
      a1.u[v + 4] = sA[p][ar0 + 16][8 + g * 4 + v];
    }
#pragma unroll
    for (int v = 0; v < 8; ++v) {
      bg.u[v] = sBg[p][ni * 16 + l][(v >> 2) * 8 + g * 4 + (v & 3)];  // K = (v>>2)*16 + g*8 + 2(v&3)
      bu.u[v] = sBu[p][ni * 16 + l][(v >> 2) * 8 + g * 4 + (v & 3)];
    }
    accg0 = __builtin_amdgcn_wmma_f32_16x16x32_bf16(false, a0.v, false, bg.v,
                                                    (short)0, accg0, false, false);
    accu0 = __builtin_amdgcn_wmma_f32_16x16x32_bf16(false, a0.v, false, bu.v,
                                                    (short)0, accu0, false, false);
    accg1 = __builtin_amdgcn_wmma_f32_16x16x32_bf16(false, a1.v, false, bg.v,
                                                    (short)0, accg1, false, false);
    accu1 = __builtin_amdgcn_wmma_f32_16x16x32_bf16(false, a1.v, false, bu.v,
                                                    (short)0, accu1, false, false);
  }

  // epilogue: act = silu(g) * u.  D layout: VGPR r -> M = r + 8*g, N = l.
#pragma unroll
  for (int s = 0; s < 2; ++s) {
    const v8f& cg = s ? accg1 : accg0;
    const v8f& cu = s ? accu1 : accu0;
#pragma unroll
    for (int r = 0; r < 8; ++r) {
      float gv = cg[r], uv = cu[r];
      float av = (gv / (1.f + __expf(-gv))) * uv;
      int m = m0 + mi * 32 + s * 16 + g * 8 + r;
      int n = n0 + ni * 16 + l;
      act[(long)m * N + n] = f2bf(av);
    }
  }
}

// =====================================================================
// Kernel 5: down-projection GEMM (act bf16 x Wdown) + scatter.
// Block 64x64, wave 32x16 -> 2 WMMA per K-step.
// A tile staged by the Tensor Data Mover (TDM) when available:
//   - pad_interval=16 dwords + pad_amount=2 dwords reproduces the 18-dword
//     LDS row stride; tensor_dim1 = nrows-m0 zero-fills OOB rows in HW.
// rowsAll==null -> shared expert: plain store (runs first on stream)
// rowsAll!=null -> routed: weighted global_atomic_add_f32 (runs second)
// =====================================================================
__global__ __launch_bounds__(256)
void down_kernel(const unsigned short* __restrict__ actBase,  // bf16 (nrows x KDIM)
                 const int*   __restrict__ rowsAll,
                 const float* __restrict__ rwtAll,
                 const int*   __restrict__ counts,
                 int nrowsFixed,
                 const float* __restrict__ BdBase,             // (KDIM x HD) per expert
                 int KDIM, long bStride, long actStride, int rowsStride,
                 float* __restrict__ out) {
  const int e = blockIdx.z;
  const float*          Bd    = BdBase  + (long)e * bStride;
  const unsigned short* act   = actBase + (long)e * actStride;
  const int*   erows = rowsAll ? rowsAll + (long)e * rowsStride : nullptr;
  const float* erwt  = rwtAll  ? rwtAll  + (long)e * rowsStride : nullptr;
  const int    nrows = counts ? imin(counts[e], rowsStride) : nrowsFixed;
  const int m0 = blockIdx.y * 64;
  if (m0 >= nrows) return;
  const int n0 = blockIdx.x * 64;

  __shared__ unsigned int sA[2][64][18];
  __shared__ unsigned int sB[2][64][18];

  const int tid  = threadIdx.x;
  const int lane = tid & 31, wave = tid >> 5;
  const int mi = wave >> 2, ni = wave & 3;
  const int l  = lane & 15, g = lane >> 4;

#if !HAVE_TDM
  // fallback path: per-thread copy of the bf16 A tile
  const int rA = tid >> 3, jA = tid & 7;        // bf16 cols jA*4..+3
  const unsigned short* aSrc0 =
      (m0 + rA      < nrows) ? act + (long)(m0 + rA     ) * KDIM + jA * 4 : nullptr;
  const unsigned short* aSrc1 =
      (m0 + rA + 32 < nrows) ? act + (long)(m0 + rA + 32) * KDIM + jA * 4 : nullptr;
#endif
  // B: fp32 (KDIM x HD)
  const int nB = (tid & 15) * 4;
  const int kB = (tid >> 4) * 2;
  const int kp = kB >> 1;
  const float* bSrc = Bd + (long)kB * HD + n0 + nB;

  auto stageA = [&](int p, int kt) {
#if HAVE_TDM
    if (wave == 0) {
      // --- Tensor DMA descriptor (D#) per CDNA5 ISA ch.8 ---
      unsigned long long ga =
          (unsigned long long)(const void*)(act + (long)m0 * KDIM + kt);
      unsigned lds = (unsigned)(unsigned long long)(const void*)&sA[p][0][0];
      u32x4 g0 = {
        1u,                                            // count=1 (user mode)
        lds,                                           // lds_addr (bytes)
        (unsigned)ga,                                  // global_addr[31:0]
        (unsigned)((ga >> 32) & 0x01FFFFFFu) | (2u << 30)  // [56:32] | type=2
      };
      unsigned td0 = (unsigned)KDIM;                   // tensor_dim0 (elements)
      unsigned td1 = (unsigned)(nrows - m0);           // tensor_dim1 (rows; HW zero-fills OOB)
      i32x8 g1 = {
        (int)((1u << 16) | (1u << 20) | (3u << 22) | (1u << 25)),
        //      data=2B    pad_en      pad_iv=16dw    pad_amt=2dw ; wg_mask=0
        (int)((td0 & 0xFFFFu) << 16),                  // abar=0 | dim0.lo
        (int)(((td0 >> 16) & 0xFFFFu) | ((td1 & 0xFFFFu) << 16)),
        (int)(((td1 >> 16) & 0xFFFFu) | (32u << 16)),  // dim1.hi | tile_dim0=32
        (int)64u,                                      // tile_dim1=64, tile_dim2=0
        (int)td0,                                      // tensor_dim0_stride[31:0]
        0, 0                                           // stride hi / dim1_stride
      };
      i32x4 g2 = { 0, 0, 0, 0 };
      i32x4 g3 = { 0, 0, 0, 0 };
      i32x8 g4 = { 0, 0, 0, 0, 0, 0, 0, 0 };  // 6-arg toolchain: extra group
      __builtin_amdgcn_tensor_load_to_lds(g0, g1, g2, g3, g4, 0);
    }
#else
    uint2 d0 = make_uint2(0u, 0u), d1 = make_uint2(0u, 0u);
    if (aSrc0) d0 = *(const uint2*)(aSrc0 + kt);
    if (aSrc1) d1 = *(const uint2*)(aSrc1 + kt);
    *(uint2*)&sA[p][rA     ][jA * 2] = d0;
    *(uint2*)&sA[p][rA + 32][jA * 2] = d1;
#endif
  };

  auto stageB = [&](int p, int kt) {
    const float* r0 = bSrc + (long)kt * HD;
    float4 b0 = *(const float4*)r0;
    float4 b1 = *(const float4*)(r0 + HD);
    sB[p][nB + 0][kp] = pack2bf(b0.x, b1.x);
    sB[p][nB + 1][kp] = pack2bf(b0.y, b1.y);
    sB[p][nB + 2][kp] = pack2bf(b0.z, b1.z);
    sB[p][nB + 3][kp] = pack2bf(b0.w, b1.w);
    if (kt + 64 < KDIM)
      __builtin_prefetch(bSrc + (long)(kt + 64) * HD, 0, 1);
  };

  v8f acc0 = {}, acc1 = {};
  stageA(0, 0);
  stageB(0, 0);
  int p = 0;
  for (int kt = 0; kt < KDIM; kt += 32, p ^= 1) {
#if HAVE_TDM
    if (wave == 0) wait_tensorcnt0();  // TDM done before barrier publishes buf
#endif
    __syncthreads();
    if (kt + 32 < KDIM) { stageA(p ^ 1, kt + 32); stageB(p ^ 1, kt + 32); }

    FragB a0, a1, b;
    const int ar0 = mi * 32 + l;
#pragma unroll
    for (int v = 0; v < 4; ++v) {
      a0.u[v]     = sA[p][ar0     ][g * 4 + v];
      a0.u[v + 4] = sA[p][ar0     ][8 + g * 4 + v];
      a1.u[v]     = sA[p][ar0 + 16][g * 4 + v];
      a1.u[v + 4] = sA[p][ar0 + 16][8 + g * 4 + v];
    }
#pragma unroll
    for (int v = 0; v < 8; ++v)
      b.u[v] = sB[p][ni * 16 + l][(v >> 2) * 8 + g * 4 + (v & 3)];

    acc0 = __builtin_amdgcn_wmma_f32_16x16x32_bf16(false, a0.v, false, b.v,
                                                   (short)0, acc0, false, false);
    acc1 = __builtin_amdgcn_wmma_f32_16x16x32_bf16(false, a1.v, false, b.v,
                                                   (short)0, acc1, false, false);
  }

#pragma unroll
  for (int s = 0; s < 2; ++s) {
    const v8f& c = s ? acc1 : acc0;
#pragma unroll
    for (int r = 0; r < 8; ++r) {
      int m = m0 + mi * 32 + s * 16 + g * 8 + r;
      int n = n0 + ni * 16 + l;
      if (m < nrows) {
        if (erows) {
          int t = erows[m];
          float val = c[r] * erwt[m];
          __hip_atomic_fetch_add(&out[(long)t * HD + n], val,
                                 __ATOMIC_RELAXED, __HIP_MEMORY_SCOPE_AGENT);
        } else {
          out[(long)m * HD + n] = c[r];   // shared expert: first writer of out
        }
      }
    }
  }
}

// =====================================================================
// Launch
// =====================================================================
extern "C" void kernel_launch(void* const* d_in, const int* in_sizes, int n_in,
                              void* d_out, int out_size, void* d_ws, size_t ws_size,
                              hipStream_t stream) {
  (void)in_sizes; (void)n_in; (void)out_size; (void)ws_size;
  const float* x       = (const float*)d_in[0];   // T x H
  const float* gate_w  = (const float*)d_in[1];   // E x H
  const float* w_gate  = (const float*)d_in[2];   // E x H x I
  const float* w_up    = (const float*)d_in[3];   // E x H x I
  const float* w_down  = (const float*)d_in[4];   // E x I x H
  const float* sw_gate = (const float*)d_in[5];   // H x Is
  const float* sw_up   = (const float*)d_in[6];   // H x Is
  const float* sw_down = (const float*)d_in[7];   // Is x H
  float* out = (float*)d_out;                     // T x H

  char* ws = (char*)d_ws;
  size_t off = 0;
  auto bump = [&](size_t b) { size_t o = off; off = (off + b + 255) & ~(size_t)255; return o; };
  int*            tk_idx = (int*)           (ws + bump((size_t)NT * TOPK * 4));
  float*          tk_wt  = (float*)         (ws + bump((size_t)NT * TOPK * 4));
  int*            counts = (int*)           (ws + bump((size_t)NE * 4));
  int*            rows   = (int*)           (ws + bump((size_t)NE * CAPE * 4));
  float*          rwt    = (float*)         (ws + bump((size_t)NE * CAPE * 4));
  unsigned short* act_r  = (unsigned short*)(ws + bump((size_t)NE * CAPE * ID * 2));
  unsigned short* act_s  = (unsigned short*)(ws + bump((size_t)NT * ISH * 2));

  gate_kernel<<<NT / 4, 128, 0, stream>>>(x, gate_w, tk_idx, tk_wt);
  zero_counts_kernel<<<1, 32, 0, stream>>>(counts);
  route_build_kernel<<<(NT * TOPK) / 256, 256, 0, stream>>>(tk_idx, tk_wt, counts, rows, rwt);
  gateup_kernel<<<dim3(ID / 64, CAPE / 64, NE), 256, 0, stream>>>(
      x, rows, counts, 0, w_gate, w_up, ID, (long)HD * ID, act_r, (long)CAPE * ID, CAPE, HD);
  gateup_kernel<<<dim3(ISH / 64, NT / 64, 1), 256, 0, stream>>>(
      x, nullptr, nullptr, NT, sw_gate, sw_up, ISH, 0, act_s, 0, 0, HD);
  down_kernel<<<dim3(HD / 64, NT / 64, 1), 256, 0, stream>>>(
      act_s, nullptr, nullptr, nullptr, NT, sw_down, ISH, 0, 0, 0, out);
  down_kernel<<<dim3(HD / 64, CAPE / 64, NE), 256, 0, stream>>>(
      act_r, rows, rwt, counts, 0, w_down, ID, (long)ID * HD, (long)CAPE * ID, CAPE, out);
}
